// VectorizedLateralEI_11519102288264
// MI455X (gfx1250) — matrix operand order repaired and checked by reading, not compiled, not run on hardware
//
#include <hip/hip_runtime.h>

// ---------------------------------------------------------------------------
// Fused lateral-EI kernel for MI455X (gfx1250, wave32, WMMA + async-to-LDS).
//
//   K[i,j] = 0.8*exp(-d2/0.72) - 1.0*exp(-d2/2.88),  d2 = |z_i - z_j|^2 (dim 8)
//   out    = 0.05 * (K / (rowsum(K)+1e-6)) @ h
//
// Pass 1 (pack_h_frag): pack h once to bf16 row-pairs in a pre-fragmented
//   layout hpf[jblk][col][k2] (8 MB ws, L2-resident): per column, the 16
//   u32 pairs of one 32-row K-block are contiguous (64 B) -> the main loop
//   stages the h tile with global_load_async_to_lds_b128 (ASYNCcnt DMA,
//   no VGPR round trip).
// Pass 2 (ei_fused_wmma): workgroup owns 32 output rows x all 512 columns;
//   loops the 8192 K-dim in steps of 32. K tile generated on the fly in f32
//   (v_exp_f32), row-summed in f32, packed to bf16 in WMMA fragment layout,
//   fed to v_wmma_f32_16x16x32_bf16 with f32 accumulation. K never touches
//   memory (saves 2x256 MB of HBM traffic).
// ---------------------------------------------------------------------------

typedef __attribute__((ext_vector_type(16))) __bf16 v16bf;
typedef __attribute__((ext_vector_type(8)))  float  v8f;

// exp(-d2/(2*sigma^2)) = exp2(d2 * (-1/(2*sigma^2*ln2)))
#define AE (-2.0037432f)    // sigma_e = 0.6
#define AI (-0.50093580f)   // sigma_i = 1.2
#define KE 0.8f
#define GAINC 0.05f
#define EPSC 1e-6f

union FragBF { int4 q[2]; v16bf v; };

__device__ __forceinline__ unsigned pack_bf16(float a, float b) {
    unsigned ua = __float_as_uint(a), ub = __float_as_uint(b);
    ua = (ua + 0x7FFFu + ((ua >> 16) & 1u)) >> 16;           // RNE
    ub = (ub + 0x7FFFu + ((ub >> 16) & 1u)) >> 16;
    return ua | (ub << 16);
}

__device__ __forceinline__ float k_eval(const float zi[8], const float* __restrict__ zj) {
    float4 a = *(const float4*)(zj);
    float4 b = *(const float4*)(zj + 4);
    float d0 = zi[0] - a.x, d1 = zi[1] - a.y, d2 = zi[2] - a.z, d3 = zi[3] - a.w;
    float d4 = zi[4] - b.x, d5 = zi[5] - b.y, d6 = zi[6] - b.z, d7 = zi[7] - b.w;
    float s = d0 * d0;
    s = fmaf(d1, d1, s); s = fmaf(d2, d2, s); s = fmaf(d3, d3, s);
    s = fmaf(d4, d4, s); s = fmaf(d5, d5, s); s = fmaf(d6, d6, s); s = fmaf(d7, d7, s);
    return KE * __builtin_amdgcn_exp2f(s * AE) - __builtin_amdgcn_exp2f(s * AI);
}

// ---- Pass 1: hpf[(jblk*N + col)*16 + k2] = pack(h[jblk*32+2k2][col], +1) ---
// LDS transpose keeps both global read and global write coalesced.
extern "C" __global__ void __launch_bounds__(256)
pack_h_frag(const float* __restrict__ h, unsigned* __restrict__ hpf, int N) {
    __shared__ float t[32][65];                  // +1 pad: conflict-free columns
    const int tid   = threadIdx.x;
    const int jbase = blockIdx.y * 32;
    const int cbase = blockIdx.x * 64;
#pragma unroll
    for (int k = 0; k < 8; ++k) {                // 32x64 f32 tile, coalesced rows
        int e = tid + k * 256;
        int r = e >> 6, c = e & 63;
        t[r][c] = h[(jbase + r) * N + cbase + c];
    }
    __syncthreads();
#pragma unroll
    for (int k = 0; k < 4; ++k) {                // write coalesced in (col,k2)
        int e = tid + k * 256;
        int k2 = e & 15, c = (e >> 4) & 63;
        unsigned val = pack_bf16(t[2 * k2][c], t[2 * k2 + 1][c]);
        hpf[((size_t)blockIdx.y * N + (cbase + c)) * 16 + k2] = val;
    }
}

// ---- Pass 2: fused K-gen + WMMA GEMM + row-sum normalization ---------------
// Block: 512 threads = 16 waves. Tile: BM=32 x BN=512, BK=32.
// LDS rows padded to 20 dwords: 16B-aligned b128 fragment loads.
template <bool USEHP>
__global__ void __launch_bounds__(512, 1)
ei_fused_wmma(const float* __restrict__ z, const float* __restrict__ h,
              const unsigned* __restrict__ hpf, float* __restrict__ out,
              int B, int N) {
    __shared__ __align__(16) float    zj_s[32 * 8];       // 1 KB  : z rows of j-tile
    __shared__ __align__(16) unsigned A2t[32 * 20];       // 2.5 KB: K tile (bf16 pairs)
    __shared__ __align__(16) unsigned H2t[512 * 20];      // 40 KB : h tile (bf16 pairs)
    __shared__ float Spart[512];
    __shared__ float Srow[32];

    const int tid  = threadIdx.x;
    const int lane = tid & 31;
    const int w    = tid >> 5;          // wave 0..15
    const int hlf  = lane >> 4;         // lane half (wmma layout)
    const int l15  = lane & 15;
    const int m0   = blockIdx.x * 32;

    // K-generation ownership: row i = tid>>4 (0..31), pair k2 = tid&15
    const int ig  = tid >> 4;
    const int k2g = tid & 15;

    float zi[8];
#pragma unroll
    for (int d = 0; d < 8; ++d) zi[d] = z[(m0 + ig) * 8 + d];

    // wave tiling of 32x512 output: msub = w&1, columns (w>>1)*64 .. +63
    const int msub  = w & 1;
    const int nbase = (w >> 1) * 64;

    // per-lane LDS byte address of this thread's H2t row (async dst)
    const unsigned h2t_lds = (unsigned)(size_t)(void*)&H2t[tid * 20];

    v8f acc[4] = {};
    float rsum = 0.0f;

    for (int j0 = 0; j0 < B; j0 += 32) {
        __syncthreads();                         // WAR vs previous iteration reads

        // ---- stage h tile: async DMA global -> LDS (ASYNCcnt), 64 B/lane ----
        if (USEHP) {
            const unsigned* gbase = hpf + (size_t)(j0 >> 5) * N * 16;
#pragma unroll
            for (int q = 0; q < 4; ++q) {
                unsigned ldst = h2t_lds + q * 16;
                unsigned goff = (unsigned)(tid * 64 + q * 16);   // bytes
                asm volatile("global_load_async_to_lds_b128 %0, %1, %2"
                             :: "v"(ldst), "v"(goff), "s"(gbase) : "memory");
            }
        } else {
            // fallback: convert in-loop (thread owns column n = tid)
#pragma unroll
            for (int k2 = 0; k2 < 16; ++k2) {
                int row = j0 + 2 * k2;
                float a = h[row * N + tid];
                float b = h[row * N + N + tid];
                H2t[tid * 20 + k2] = pack_bf16(a, b);
            }
        }

        // ---- stage z_j tile (256 contiguous floats) ----
        if (tid < 256) zj_s[tid] = z[j0 * 8 + tid];

        if (USEHP) asm volatile("s_wait_asynccnt 0x0" ::: "memory");
        __syncthreads();                         // zj_s + H2t ready

        // ---- prefetch B fragments now (H2t ready); overlaps K-gen VALU ----
        // B 32x16 bf16, this code's contiguous k order: vgpr v holds k = 2v + 16*half => k2 in {8h..8h+7}
        FragBF bfr[4];
#pragma unroll
        for (int s = 0; s < 4; ++s) {
            const int col = nbase + s * 16 + l15;
            bfr[s].q[0] = *(const int4*)&H2t[col * 20 + 8 * hlf];
            bfr[s].q[1] = *(const int4*)&H2t[col * 20 + 8 * hlf + 4];
        }

        // ---- generate K tile [32 x 32] in f32, pack to bf16, row-sum ----
        {
            int jl = 2 * k2g;
            float kv0 = k_eval(zi, zj_s + jl * 8);
            float kv1 = k_eval(zi, zj_s + jl * 8 + 8);
            rsum += kv0 + kv1;
            A2t[ig * 20 + k2g] = pack_bf16(kv0, kv1);
        }
        __syncthreads();                         // A2t ready

        // ---- A fragment + 4 back-to-back WMMAs ----
        // A 16x32 bf16, this code's contiguous k order: vgpr v holds k = 2v + 16*half (matches B)
        FragBF afr;
        const int arow = msub * 16 + l15;
        afr.q[0] = *(const int4*)&A2t[arow * 20 + 8 * hlf];
        afr.q[1] = *(const int4*)&A2t[arow * 20 + 8 * hlf + 4];
#pragma unroll
        for (int s = 0; s < 4; ++s) {
            acc[s] = __builtin_amdgcn_wmma_f32_16x16x32_bf16(
                false, afr.v, false, bfr[s].v, (short)0, acc[s], false, false);
        }
    }

    // ---- reduce row sums (f32), fold in GAIN and EPS ----
    Spart[tid] = rsum;
    __syncthreads();
    if (tid < 32) {
        float s = 0.0f;
#pragma unroll
        for (int g = 0; g < 16; ++g) s += Spart[tid * 16 + g];
        Srow[tid] = GAINC / (s + EPSC);
    }
    __syncthreads();

    // ---- scale + store: C/D layout: vgpr v -> row v + 8*half, col = lane&15 ----
#pragma unroll
    for (int v = 0; v < 8; ++v) {
        const int mrow = msub * 16 + v + 8 * hlf;
        const float sc = Srow[mrow];
        float* orow = out + (m0 + mrow) * N + nbase + l15;
        orow[0]  = acc[0][v] * sc;
        orow[16] = acc[1][v] * sc;
        orow[32] = acc[2][v] * sc;
        orow[48] = acc[3][v] * sc;
    }
}

extern "C" void kernel_launch(void* const* d_in, const int* in_sizes, int n_in,
                              void* d_out, int out_size, void* d_ws, size_t ws_size,
                              hipStream_t stream) {
    const float* z = (const float*)d_in[0];
    const float* h = (const float*)d_in[1];
    float* out = (float*)d_out;
    const int B = in_sizes[0] / 8;      // 8192
    const int N = in_sizes[1] / B;      // 512
    const size_t hp_bytes = (size_t)(B / 2) * N * sizeof(unsigned);   // 8 MB

    dim3 grid(B / 32);                  // 256 workgroups x 16 waves

    if (ws_size >= hp_bytes && d_ws != nullptr) {
        unsigned* hpf = (unsigned*)d_ws;
        dim3 pgrid(N / 64, B / 32);     // (8, 256) tiles of 32 rows x 64 cols
        pack_h_frag<<<pgrid, 256, 0, stream>>>(h, hpf, N);
        ei_fused_wmma<true><<<grid, 512, 0, stream>>>(z, h, hpf, out, B, N);
    } else {
        ei_fused_wmma<false><<<grid, 512, 0, stream>>>(z, h, nullptr, out, B, N);
    }
}
